// VanillaConvolutionWrapper_25082609009375
// MI455X (gfx1250) — matrix operand order repaired and checked by reading, not compiled, not run on hardware
//
#include <hip/hip_runtime.h>
#include <hip/hip_bf16.h>

typedef __bf16 bf16_t;
typedef __attribute__((ext_vector_type(16))) __bf16 v16bf;
typedef __attribute__((ext_vector_type(8)))  __bf16 v8bf;
typedef __attribute__((ext_vector_type(8)))  float  v8f;
typedef __attribute__((ext_vector_type(4)))  unsigned int u32x4;
typedef __attribute__((ext_vector_type(8)))  int          i32x8;
typedef __attribute__((ext_vector_type(4)))  int          i32x4;

#define CIN     128
#define COUT    256
#define HW      56
#define SPATIAL (HW * HW)          // 3136
#define NBATCH  32
#define NPIX    (NBATCH * SPATIAL) // 100352, multiple of 128
#define LDSPAD  40                 // 80 bytes/row: 16B aligned, conflict-free stride

#if __has_builtin(__builtin_amdgcn_tensor_load_to_lds)
#define USE_TDM 1
#else
#define USE_TDM 0
#endif

// ---------------------------------------------------------------------------
// Pass 1: repack weights OIHW f32 -> bf16, layout [tap*4+cblk][k_out=256][kl=32]
// element (tap, c=cblk*32+kl, k):  wb[((tap*4+cblk)*COUT + k)*32 + kl]
// => for a given (tap, cblk, ko) the B tile is 64 rows (n) x 64 contiguous
//    bytes (32 bf16 k-values): a clean 2D TDM tile.
// ---------------------------------------------------------------------------
__global__ __launch_bounds__(256) void conv_weight_cvt(
    const float* __restrict__ w, bf16_t* __restrict__ wb) {
  const int k    = threadIdx.x;          // 0..255 output channel
  const int blk  = blockIdx.x;           // 0..1151
  const int c    = blk / 9;              // 0..127
  const int tap  = blk - c * 9;          // 0..8
  const int cblk = c >> 5;               // 0..3
  const int kl   = c & 31;               // 0..31
  const float v  = w[((k * CIN + c) * 9) + tap];
  wb[(((tap * 4 + cblk) * COUT) + k) * 32 + kl] = (bf16_t)v;
}

// ---------------------------------------------------------------------------
// Pass 2: implicit-GEMM conv, bf16 WMMA with f32 accumulation.
// Block tile: 128 pixels (M) x 64 out-channels (N). 8 waves: 4 (M) x 2 (N),
// each wave owns a 32x32 tile => 2x2 v_wmma_f32_16x16x32_bf16 accumulators.
// B tile staged by the Tensor Data Mover (wave 0), A tile staged manually
// (needs halo gather + zero padding).
// ---------------------------------------------------------------------------
__global__ __launch_bounds__(256) void conv_wmma_bf16(
    const float*  __restrict__ x,     // [32][128][56][56]
    const bf16_t* __restrict__ wb,    // [36][256][32] bf16
    const float*  __restrict__ bias,  // [256]
    float*        __restrict__ out) { // [32][256][56][56]
  __shared__ bf16_t As[128][LDSPAD];  // pixels x k-chunk (32 valid)
  __shared__ bf16_t Bs[64][LDSPAD];   // out-ch x k-chunk (32 valid)

  const int tid  = threadIdx.x;
  const int lane = tid & 31;
  const int wave = tid >> 5;
  const int wm   = wave & 3;          // wave M position (0..3)
  const int wn   = wave >> 2;         // wave N position (0..1)
  const int lh   = lane & 15;
  const int sel  = lane >> 4;         // 0: lanes 0-15, 1: lanes 16-31

  const int mbase = blockIdx.x * 128; // pixel base
  const int ko    = blockIdx.y * 64;  // out-channel base

  // ---- hoisted pixel decode for the 16 A-staging slots (k-loop invariant) --
  int bi[16], hv[16], wv[16];
#pragma unroll
  for (int i = 0; i < 16; ++i) {
    const int e    = tid + i * 256;     // 0..4095
    const int ml   = e & 127;           // pixel within tile
    const int cl   = e >> 7;            // channel within chunk (0..31)
    const int pix  = mbase + ml;
    const int nimg = pix / SPATIAL;
    const int rem  = pix - nimg * SPATIAL;
    hv[i] = rem / HW;
    wv[i] = rem - hv[i] * HW;
    bi[i] = (nimg * CIN + cl) * SPATIAL + rem;   // add cb*SPATIAL + doff later
  }

  v8f acc[2][2] = {};                 // f32 accumulators

  for (int tap = 0; tap < 9; ++tap) {
    const int dr   = tap / 3 - 1;
    const int dc   = tap % 3 - 1;
    const int doff = dr * HW + dc;
    for (int cb = 0; cb < CIN; cb += 32) {
      const int cbase = cb * SPATIAL;
      const int tap4  = tap * 4 + (cb >> 5);

      // ---- stage A: 128 pixels x 32 channels, f32 -> bf16, zero padding ----
#pragma unroll
      for (int i = 0; i < 16; ++i) {
        const int e  = tid + i * 256;
        const int ml = e & 127;
        const int cl = e >> 7;
        float v = 0.0f;
        if ((unsigned)(hv[i] + dr) < (unsigned)HW &&
            (unsigned)(wv[i] + dc) < (unsigned)HW)
          v = x[bi[i] + cbase + doff];
        As[ml][cl] = (bf16_t)v;
      }

      // ---- stage B: 64 out-channels x 32 k-values ----
#if USE_TDM
      if (wave == 0) {
        const unsigned long long ga =
            (unsigned long long)(uintptr_t)wb +
            (unsigned long long)(tap4 * COUT + ko) * 64ull;
        const unsigned int lds_b = (unsigned int)(uintptr_t)&Bs[0][0];
        u32x4 g0;
        g0[0] = 1u;                                    // count=1, user mode
        g0[1] = lds_b;                                 // lds_addr (bytes)
        g0[2] = (unsigned int)ga;                      // global_addr[31:0]
        g0[3] = (unsigned int)((ga >> 32) & 0x1FFFFFFu) | (2u << 30); // type=2
        i32x8 g1;
        g1[0] = (1 << 16)      // data_size = 2 bytes
              | (1 << 20)      // pad_enable
              | (3 << 22)      // pad_interval: 16 dwords (= 64B row)
              | (3 << 25);     // pad_amount: 4 dwords (= 16B pad -> 80B stride)
        g1[1] = (32 << 16);    // tensor_dim0 = 32 (low 16 in bits 63:48)
        g1[2] = (64 << 16);    // tensor_dim0 hi = 0; tensor_dim1 = 64
        g1[3] = (32 << 16);    // tensor_dim1 hi = 0; tile_dim0 = 32
        g1[4] = 64;            // tile_dim1 = 64; tile_dim2 = 0
        g1[5] = 32;            // tensor_dim0_stride = 32 elements
        g1[6] = 0;
        g1[7] = 0;
        const i32x4 z4 = {0, 0, 0, 0};
#if defined(__clang_major__) && __clang_major__ >= 23
        const i32x8 z8 = {0, 0, 0, 0, 0, 0, 0, 0};
        __builtin_amdgcn_tensor_load_to_lds(g0, g1, z4, z4, z8, 0);
#else
        __builtin_amdgcn_tensor_load_to_lds(g0, g1, z4, z4, 0);
#endif
        __builtin_amdgcn_s_wait_tensorcnt(0);
      }
#else
#pragma unroll
      for (int i = 0; i < 8; ++i) {
        const int e  = tid + i * 256;   // 0..2047
        const int kl = e & 31;
        const int nl = e >> 5;
        Bs[nl][kl] = wb[((tap4 * COUT) + ko + nl) * 32 + kl];
      }
#endif
      __syncthreads();

      // ---- build fragments per ISA 16-bit A/B VGPR layouts ----
      v16bf afrag[2], bfrag[2];
#pragma unroll
      for (int i = 0; i < 2; ++i) {
        // A row m = lane&15; lanes 0-15 hold K {0..7,16..23}, lanes 16-31 rest
        const bf16_t* pa = &As[wm * 32 + i * 16 + lh][sel * 8];
        const v8bf lo = *(const v8bf*)pa;        // K koff..koff+7
        const v8bf hi = *(const v8bf*)(pa + 16); // K koff+16..koff+23
        afrag[i] = __builtin_shufflevector(lo, hi, 0, 1, 2, 3, 4, 5, 6, 7,
                                           8, 9, 10, 11, 12, 13, 14, 15);
      }
#pragma unroll
      for (int j = 0; j < 2; ++j) {
        // B col n = lane&15; lanes 0-15 hold K {0..7,16..23}, lanes 16-31 rest
        const bf16_t* pb = &Bs[wn * 32 + j * 16 + lh][sel * 8];
        const v8bf lo = *(const v8bf*)pb;
        const v8bf hi = *(const v8bf*)(pb + 16);
        bfrag[j] = __builtin_shufflevector(lo, hi, 0, 1, 2, 3, 4, 5, 6, 7,
                                           8, 9, 10, 11, 12, 13, 14, 15);
      }

      // ---- 4 WMMAs per wave ----
#pragma unroll
      for (int i = 0; i < 2; ++i)
#pragma unroll
        for (int j = 0; j < 2; ++j)
          acc[i][j] = __builtin_amdgcn_wmma_f32_16x16x32_bf16(
              false, afrag[i], false, bfrag[j], (short)0, acc[i][j],
              false, false);

      __syncthreads();
    }
  }

  // ---- epilogue: add bias, scatter per 16x16 f32 C/D layout ----
#pragma unroll
  for (int j = 0; j < 2; ++j) {
    const int   kk = ko + wn * 32 + j * 16 + lh;   // N = lane&15
    const float bv = bias[kk];
#pragma unroll
    for (int i = 0; i < 2; ++i) {
#pragma unroll
      for (int r = 0; r < 8; ++r) {
        // VGPR r holds M=r (lanes 0-15) or M=r+8 (lanes 16-31)
        const int pix  = mbase + wm * 32 + i * 16 + r + sel * 8;
        const int nimg = pix / SPATIAL;
        const int rem  = pix - nimg * SPATIAL;
        out[(nimg * COUT + kk) * SPATIAL + rem] = acc[i][j][r] + bv;
      }
    }
  }
}

// ---------------------------------------------------------------------------
extern "C" void kernel_launch(void* const* d_in, const int* in_sizes, int n_in,
                              void* d_out, int out_size, void* d_ws,
                              size_t ws_size, hipStream_t stream) {
  const float* x    = (const float*)d_in[0];
  const float* w    = (const float*)d_in[1];
  const float* bias = (const float*)d_in[2];
  float*       out  = (float*)d_out;
  bf16_t*      wb   = (bf16_t*)d_ws;   // needs 36*256*32*2 = 576 KiB

  // repack weights to bf16 TDM-tileable layout
  conv_weight_cvt<<<CIN * 9, 256, 0, stream>>>(w, wb);

  // main conv: 784 M-tiles x 4 N-tiles
  dim3 grid(NPIX / 128, COUT / 64);
  conv_wmma_bf16<<<grid, 256, 0, stream>>>(x, wb, bias, out);
}
